// MultiHeadAttention_42150809043692
// MI455X (gfx1250) — matrix operand order repaired and checked by reading, not compiled, not run on hardware
//
#include <hip/hip_runtime.h>

// ---------------- problem constants ----------------
#define D_MODEL 1024
#define SEQ     2048
#define BATCH   4
#define HEADS   16
#define HDIM    64
#define M_ROWS  (BATCH * SEQ)      // 8192
#define EL_IN   ((size_t)M_ROWS * D_MODEL)   // 8388608
#define EL_W    ((size_t)D_MODEL * D_MODEL)  // 1048576
#define KBLK    64                 // attention key-block
#define NIT     (SEQ / KBLK)       // 32 iterations

typedef __attribute__((ext_vector_type(16))) __bf16       v16bf;
typedef __attribute__((ext_vector_type(8)))  float        v8f;
typedef __attribute__((ext_vector_type(8)))  unsigned int v8u;
typedef __attribute__((ext_vector_type(4)))  unsigned int uint4v;
typedef __attribute__((ext_vector_type(2)))  unsigned int uint2v;
typedef __attribute__((ext_vector_type(4)))  float        float4v;

// fp32 -> bf16, round to nearest even
__device__ __forceinline__ unsigned short f2bf(float f) {
  union { float f; unsigned u; } v; v.f = f;
  unsigned r = v.u + 0x7FFFu + ((v.u >> 16) & 1u);
  return (unsigned short)(r >> 16);
}
__device__ __forceinline__ unsigned pack2(float a, float b) {
  return (unsigned)f2bf(a) | ((unsigned)f2bf(b) << 16);
}
__device__ __forceinline__ v16bf as_bf16x16(v8u u) {
  return __builtin_bit_cast(v16bf, u);
}
// A-matrix 16x32 bf16 layout: lane group g, VGPR j -> K pair start
__device__ __forceinline__ int kkA(int j, int g) {
  return ((j & 4) << 2) + (g << 3) + ((j & 3) << 1);
}
// B-matrix 32x16 bf16 layout: lane group g, VGPR j -> K pair start
__device__ __forceinline__ int kkB(int j, int g) {
  return ((j & 4) << 2) + (g << 3) + ((j & 3) << 1);
}
#define WMMA_BF16(a, b, c) \
  __builtin_amdgcn_wmma_f32_16x16x32_bf16(false, (a), false, (b), (short)0, (c), false, false)

// ---- Tensor Data Mover availability ----
#if defined(__has_builtin)
#if __has_builtin(__builtin_amdgcn_tensor_load_to_lds) && \
    __has_builtin(__builtin_amdgcn_s_wait_tensorcnt)
#define USE_TDM 1
#endif
#endif
#ifndef USE_TDM
#define USE_TDM 0
#endif

#if USE_TDM
typedef __attribute__((ext_vector_type(4))) unsigned int tdm_g0_t;
typedef __attribute__((ext_vector_type(8))) int          tdm_g1_t;
typedef __attribute__((ext_vector_type(4))) int          tdm_g2_t;
typedef __attribute__((ext_vector_type(8))) int          tdm_g4_t;

// Build a D# (groups 0/1 per CDNA5 ISA §8.3/8.4) and issue a 2-D tile load.
// data_size = 2 bytes (bf16). tile_d1 == 0 -> 1-D copy of tile_d0 elements.
// 6-arg builtin form (clang-23 / therock-10.0 headers).
__device__ __forceinline__ void tdm_load_2d(unsigned lds_off, const void* gaddr,
                                            unsigned tile_d0, unsigned tile_d1,
                                            unsigned tensor_d0, unsigned tensor_d1,
                                            unsigned stride0_elems) {
  const unsigned long long ga = (unsigned long long)gaddr;
  tdm_g0_t g0;
  g0[0] = 1u;                                             // count=1, user descriptor
  g0[1] = lds_off;                                        // lds_addr [63:32]
  g0[2] = (unsigned)(ga & 0xFFFFFFFFu);                   // global_addr [95:64]
  g0[3] = (unsigned)((ga >> 32) & 0x1FFFFFFu) | (2u << 30); // addr[56:32] | type=2
  tdm_g1_t g1;
  g1[0] = (int)(1u << 16);                                // data_size=1 (2 bytes)
  g1[1] = (int)((tensor_d0 & 0xFFFFu) << 16);             // tensor_dim0[15:0]
  g1[2] = (int)(((tensor_d0 >> 16) & 0xFFFFu) | ((tensor_d1 & 0xFFFFu) << 16));
  g1[3] = (int)(((tensor_d1 >> 16) & 0xFFFFu) | ((tile_d0 & 0xFFFFu) << 16));
  g1[4] = (int)(tile_d1 & 0xFFFFu);                       // tile_dim1, tile_dim2=0
  g1[5] = (int)stride0_elems;                             // tensor_dim0_stride[31:0]
  g1[6] = 0;
  g1[7] = 0;
  const tdm_g2_t z4 = {0, 0, 0, 0};
  const tdm_g4_t z8 = {0, 0, 0, 0, 0, 0, 0, 0};
  __builtin_amdgcn_tensor_load_to_lds(g0, g1, z4, z4, z8, 0);
}
#endif

// =====================================================================
// Kernel 0: fp32 -> bf16 bulk convert (4 elements / thread)
// =====================================================================
__global__ void __launch_bounds__(256) cvt_bf16_kernel(
    const float* __restrict__ in, unsigned short* __restrict__ out, int n4) {
  const int i = blockIdx.x * 256 + threadIdx.x;
  if (i >= n4) return;
  const float4v v = ((const float4v*)in)[i];
  uint2v u;
  u[0] = pack2(v[0], v[1]);
  u[1] = pack2(v[2], v[3]);
  ((uint2v*)out)[i] = u;
}

// =====================================================================
// Kernel 1: GEMM  out = A @ W^T + bias   (bf16 A, bf16 W, 32x32 wave tile)
// mode 0: bf16 out [B,H,S,Dh]   mode 2: bf16 out [B,H,Dh,S]   mode 3: f32 row-major
// =====================================================================
__global__ void __launch_bounds__(128) gemm_kernel(
    const unsigned short* __restrict__ Abf,  // [M_ROWS, D_MODEL]
    const unsigned short* __restrict__ Wbf,  // [D_MODEL, D_MODEL]
    const float* __restrict__ bias,          // [D_MODEL]
    unsigned short* __restrict__ out_bf,
    float* __restrict__ out_f32,
    int mode)
{
  const int wave = threadIdx.x >> 5;
  const int lane = threadIdx.x & 31;
  const int g    = lane >> 4;
  const int mr   = lane & 15;

  const int tile = blockIdx.x * 4 + wave;      // 8192 tiles of 32x32
  const int mT = tile >> 5;                    // D_MODEL/32 = 32 tiles wide
  const int nT = tile & 31;
  const int m0 = mT * 32, n0 = nT * 32;

  const unsigned* a0 = (const unsigned*)(Abf + (size_t)(m0 + mr) * D_MODEL);
  const unsigned* a1 = (const unsigned*)(Abf + (size_t)(m0 + 16 + mr) * D_MODEL);
  const unsigned* b0 = (const unsigned*)(Wbf + (size_t)(n0 + mr) * D_MODEL);
  const unsigned* b1 = (const unsigned*)(Wbf + (size_t)(n0 + 16 + mr) * D_MODEL);

  v8f acc[2][2] = {};
  for (int kb = 0; kb < D_MODEL; kb += 32) {
    __builtin_prefetch(a0 + ((kb + 128) >> 1), 0, 3);
    __builtin_prefetch(a1 + ((kb + 128) >> 1), 0, 3);
    v8u au0, au1, bu0, bu1;
#pragma unroll
    for (int j = 0; j < 8; ++j) {
      const int ka = (kb + kkA(j, g)) >> 1;
      const int kw = (kb + kkB(j, g)) >> 1;
      au0[j] = a0[ka];  au1[j] = a1[ka];
      bu0[j] = b0[kw];  bu1[j] = b1[kw];
    }
    const v16bf af0 = as_bf16x16(au0), af1 = as_bf16x16(au1);
    const v16bf bf0 = as_bf16x16(bu0), bf1 = as_bf16x16(bu1);
    acc[0][0] = WMMA_BF16(af0, bf0, acc[0][0]);
    acc[0][1] = WMMA_BF16(af0, bf1, acc[0][1]);
    acc[1][0] = WMMA_BF16(af1, bf0, acc[1][0]);
    acc[1][1] = WMMA_BF16(af1, bf1, acc[1][1]);
  }

#pragma unroll
  for (int at = 0; at < 2; ++at) {
#pragma unroll
    for (int ct = 0; ct < 2; ++ct) {
      const int col = n0 + ct * 16 + mr;
      const float bv = bias[col];
      if (mode == 3) {
#pragma unroll
        for (int r = 0; r < 8; ++r) {
          const int row = m0 + at * 16 + r + 8 * g;
          out_f32[(size_t)row * D_MODEL + col] = acc[at][ct][r] + bv;
        }
      } else {
        const int h = col >> 6, d = col & (HDIM - 1);
#pragma unroll
        for (int r = 0; r < 8; ++r) {
          const int row = m0 + at * 16 + r + 8 * g;
          const int b = row >> 11;
          const int s = row & (SEQ - 1);
          const float v = acc[at][ct][r] + bv;
          size_t idx;
          if (mode == 2) idx = ((size_t)(b * HEADS + h) * HDIM + d) * SEQ + s;  // V^T
          else           idx = ((size_t)(b * HEADS + h) * SEQ + s) * HDIM + d;  // Q,K
          out_bf[idx] = f2bf(v);
        }
      }
    }
  }
}

// =====================================================================
// Kernel 2: flash attention, 64-key blocks, double-buffered TDM staging
// =====================================================================
__global__ void __launch_bounds__(128) attn_kernel(
    const unsigned short* __restrict__ Qw,   // [BH, SEQ, 64] bf16
    const unsigned short* __restrict__ Kw,   // [BH, SEQ, 64] bf16
    const unsigned short* __restrict__ Vt,   // [BH, 64, SEQ] bf16
    unsigned short* __restrict__ Ctx)        // [B, SEQ, D_MODEL] bf16
{
  __shared__ unsigned short Ktile[2][KBLK][HDIM];  // keys x headdim   (2 x 8 KB)
  __shared__ unsigned short Vtile[2][HDIM][KBLK];  // headdim x keys   (2 x 8 KB)
  __shared__ unsigned short plds[4][16][KBLK];     // per-wave P tiles (8 KB)

  const int wave = threadIdx.x >> 5;
  const int lane = threadIdx.x & 31;
  const int g    = lane >> 4;
  const int mr   = lane & 15;

  const int qTiles = SEQ / 64;               // 32
  const int bh   = blockIdx.x / qTiles;
  const int qBlk = blockIdx.x % qTiles;
  const int q0   = qBlk * 64 + wave * 16;

  // Q fragments (K = 0..31 and 32..63 of Dh), kept in registers all along
  const unsigned* qrow = (const unsigned*)(Qw + ((size_t)bh * SEQ + q0 + mr) * HDIM);
  v8u qu0, qu1;
#pragma unroll
  for (int j = 0; j < 8; ++j) {
    const int k = kkA(j, g);
    qu0[j] = qrow[k >> 1];
    qu1[j] = qrow[(32 + k) >> 1];
  }
  const v16bf qf0 = as_bf16x16(qu0), qf1 = as_bf16x16(qu1);

  float mrow[8], lrow[8];
  v8f o[4] = {{}, {}, {}, {}};
#pragma unroll
  for (int r = 0; r < 8; ++r) { mrow[r] = -1e30f; lrow[r] = 0.0f; }

  const float scale = 0.125f;                // 1/sqrt(64)

#if USE_TDM
  // prime the pipeline: stage block 0 into buffer 0
  if (wave == 0) {
    tdm_load_2d((unsigned)(unsigned long long)&Ktile[0][0][0],
                Kw + (size_t)bh * SEQ * HDIM,
                /*tile*/ KBLK * HDIM, 0, /*tensor*/ KBLK * HDIM, 1, KBLK * HDIM);
    tdm_load_2d((unsigned)(unsigned long long)&Vtile[0][0][0],
                Vt + (size_t)bh * HDIM * SEQ,
                /*tile*/ KBLK, HDIM, /*tensor*/ SEQ, HDIM, SEQ);
  }
#endif

  for (int it = 0; it < NIT; ++it) {
    const int cur = it & 1;
    const int kv  = it * KBLK;

#if USE_TDM
    if (wave == 0) {
      if (it + 1 < NIT) {
        const int nxt = cur ^ 1;
        tdm_load_2d((unsigned)(unsigned long long)&Ktile[nxt][0][0],
                    Kw + ((size_t)bh * SEQ + kv + KBLK) * HDIM,
                    KBLK * HDIM, 0, KBLK * HDIM, 1, KBLK * HDIM);
        tdm_load_2d((unsigned)(unsigned long long)&Vtile[nxt][0][0],
                    Vt + (size_t)bh * HDIM * SEQ + kv + KBLK,
                    KBLK, HDIM, SEQ, HDIM, SEQ);
        __builtin_amdgcn_s_wait_tensorcnt(2);  // in-order: oldest pair (cur) done
      } else {
        __builtin_amdgcn_s_wait_tensorcnt(0);
      }
    }
#else
    {
      // cooperative copy of current block (8 KB each, 512 x uint4, 128 threads)
      const uint4v* gk = (const uint4v*)(Kw + ((size_t)bh * SEQ + kv) * HDIM);
      uint4v* sk = (uint4v*)&Ktile[cur][0][0];
      uint4v* sv = (uint4v*)&Vtile[cur][0][0];
#pragma unroll
      for (int i = 0; i < 4; ++i) {
        const int idx = threadIdx.x + 128 * i;
        sk[idx] = gk[idx];
        const int row = idx >> 3, c = idx & 7;   // 8 x uint4 per 128 B row
        const uint4v* gv = (const uint4v*)(Vt + ((size_t)bh * HDIM + row) * SEQ + kv);
        sv[idx] = gv[c];
      }
    }
#endif
    __syncthreads();

    // ---- scores S[16x64] = Q . K^T, K fragments from LDS ----
    v8f s[4] = {{}, {}, {}, {}};
#pragma unroll
    for (int t = 0; t < 4; ++t) {
      const unsigned* kr = (const unsigned*)&Ktile[cur][t * 16 + mr][0];
      v8u k0, k1;
#pragma unroll
      for (int j = 0; j < 8; ++j) {
        const int k = kkB(j, g) >> 1;
        k0[j] = kr[k];
        k1[j] = kr[16 + k];
      }
      s[t] = WMMA_BF16(qf0, as_bf16x16(k0), s[t]);
      s[t] = WMMA_BF16(qf1, as_bf16x16(k1), s[t]);
    }

    // ---- online softmax (row stats replicated across 16-lane halves) ----
#pragma unroll
    for (int r = 0; r < 8; ++r) {
      const float x0 = s[0][r] * scale;
      const float x1 = s[1][r] * scale;
      const float x2 = s[2][r] * scale;
      const float x3 = s[3][r] * scale;
      float mx = fmaxf(fmaxf(x0, x1), fmaxf(x2, x3));
      mx = fmaxf(mx, __shfl_xor(mx, 1, 32));
      mx = fmaxf(mx, __shfl_xor(mx, 2, 32));
      mx = fmaxf(mx, __shfl_xor(mx, 4, 32));
      mx = fmaxf(mx, __shfl_xor(mx, 8, 32));
      const float mnew = fmaxf(mrow[r], mx);
      const float corr = __expf(mrow[r] - mnew);
      const float p0 = __expf(x0 - mnew);
      const float p1 = __expf(x1 - mnew);
      const float p2 = __expf(x2 - mnew);
      const float p3 = __expf(x3 - mnew);
      float ps = (p0 + p1) + (p2 + p3);
      ps += __shfl_xor(ps, 1, 32);
      ps += __shfl_xor(ps, 2, 32);
      ps += __shfl_xor(ps, 4, 32);
      ps += __shfl_xor(ps, 8, 32);
      lrow[r] = lrow[r] * corr + ps;
      mrow[r] = mnew;
      o[0][r] *= corr; o[1][r] *= corr; o[2][r] *= corr; o[3][r] *= corr;
      unsigned short* prow_st = &plds[wave][r + 8 * g][0];
      prow_st[mr]      = f2bf(p0);              // C layout -> row-major LDS
      prow_st[16 + mr] = f2bf(p1);
      prow_st[32 + mr] = f2bf(p2);
      prow_st[48 + mr] = f2bf(p3);
    }
    __syncthreads();

    // ---- reload P (16x64) as two A fragments ----
    const unsigned* prow = (const unsigned*)&plds[wave][mr][0];
    v8u pu0, pu1;
#pragma unroll
    for (int j = 0; j < 8; ++j) {
      const int k = kkA(j, g) >> 1;
      pu0[j] = prow[k];
      pu1[j] = prow[16 + k];
    }
    const v16bf pf0 = as_bf16x16(pu0), pf1 = as_bf16x16(pu1);

    // ---- O += P . V, V fragments from LDS (contiguous per head-dim row) ----
#pragma unroll
    for (int t = 0; t < 4; ++t) {
      const unsigned* vr = (const unsigned*)&Vtile[cur][t * 16 + mr][0];
      v8u v0, v1;
#pragma unroll
      for (int j = 0; j < 8; ++j) {
        const int k = kkB(j, g) >> 1;
        v0[j] = vr[k];
        v1[j] = vr[16 + k];
      }
      o[t] = WMMA_BF16(pf0, as_bf16x16(v0), o[t]);
      o[t] = WMMA_BF16(pf1, as_bf16x16(v1), o[t]);
    }
    __syncthreads();   // all reads of buffer `cur` done before it is re-staged
  }

  // ---- normalize and store context bf16 [B,S,D] ----
  const int b = bh >> 4, h = bh & 15;
#pragma unroll
  for (int r = 0; r < 8; ++r) {
    const float inv = 1.0f / lrow[r];
    const int qq = q0 + r + 8 * g;
#pragma unroll
    for (int t = 0; t < 4; ++t) {
      const int d = t * 16 + mr;
      Ctx[((size_t)b * SEQ + qq) * D_MODEL + h * HDIM + d] = f2bf(o[t][r] * inv);
    }
  }
}

// =====================================================================
extern "C" void kernel_launch(void* const* d_in, const int* in_sizes, int n_in,
                              void* d_out, int out_size, void* d_ws, size_t ws_size,
                              hipStream_t stream) {
  const float* xq = (const float*)d_in[0];
  const float* xk = (const float*)d_in[1];
  const float* xv = (const float*)d_in[2];
  const float* Wq = (const float*)d_in[3];
  const float* bq = (const float*)d_in[4];
  const float* Wk = (const float*)d_in[5];
  const float* bk = (const float*)d_in[6];
  const float* Wv = (const float*)d_in[7];
  const float* bv = (const float*)d_in[8];
  const float* Wo = (const float*)d_in[9];
  const float* bo = (const float*)d_in[10];

  // workspace layout (bf16 ushorts): 3 inputs, 4 weights, Q, K, V^T, Ctx
  unsigned short* p = (unsigned short*)d_ws;
  unsigned short* xq_bf = p;  p += EL_IN;
  unsigned short* xk_bf = p;  p += EL_IN;
  unsigned short* xv_bf = p;  p += EL_IN;
  unsigned short* wq_bf = p;  p += EL_W;
  unsigned short* wk_bf = p;  p += EL_W;
  unsigned short* wv_bf = p;  p += EL_W;
  unsigned short* wo_bf = p;  p += EL_W;
  unsigned short* qws   = p;  p += EL_IN;
  unsigned short* kws   = p;  p += EL_IN;
  unsigned short* vtws  = p;  p += EL_IN;
  unsigned short* ctx   = p;  p += EL_IN;

  auto cvt = [&](const float* src, unsigned short* dst, size_t n) {
    const int n4 = (int)(n / 4);
    cvt_bf16_kernel<<<(n4 + 255) / 256, 256, 0, stream>>>(src, dst, n4);
  };
  cvt(xq, xq_bf, EL_IN);
  cvt(xk, xk_bf, EL_IN);
  cvt(xv, xv_bf, EL_IN);
  cvt(Wq, wq_bf, EL_W);
  cvt(Wk, wk_bf, EL_W);
  cvt(Wv, wv_bf, EL_W);
  cvt(Wo, wo_bf, EL_W);

  const int gemmBlocks = (M_ROWS / 32) * (D_MODEL / 32) / 4;  // 2048 blocks x 4 waves
  gemm_kernel<<<gemmBlocks, 128, 0, stream>>>(xq_bf, wq_bf, bq, qws,  nullptr, 0);
  gemm_kernel<<<gemmBlocks, 128, 0, stream>>>(xk_bf, wk_bf, bk, kws,  nullptr, 0);
  gemm_kernel<<<gemmBlocks, 128, 0, stream>>>(xv_bf, wv_bf, bv, vtws, nullptr, 2);

  const int attnBlocks = (BATCH * HEADS) * (SEQ / 64);        // 2048
  attn_kernel<<<attnBlocks, 128, 0, stream>>>(qws, kws, vtws, ctx);

  gemm_kernel<<<gemmBlocks, 128, 0, stream>>>(ctx, wo_bf, bo, nullptr, (float*)d_out, 3);
}
